// SpectralConv1dODE_5196910428983
// MI455X (gfx1250) — matrix operand order repaired and checked by reading, not compiled, not run on hardware
//
#include <hip/hip_runtime.h>

// SpectralConv1dODE forward for MI455X (gfx1250, wave32, WMMA f16->f32, TDM).
// B=8, N=1024, E=128, T=16, M=256.

typedef __attribute__((ext_vector_type(16))) _Float16 v16h;
typedef __attribute__((ext_vector_type(8)))  _Float16 v8h;
typedef __attribute__((ext_vector_type(8)))  float    v8f;
typedef __attribute__((ext_vector_type(4)))  unsigned int v4u;
typedef __attribute__((ext_vector_type(8)))  unsigned int v8u;

#define B_ 8
#define N_ 1024
#define E_ 128
#define T_ 16
#define M_ 256

// ---------------- WMMA fragment loaders (ISA 7.12.2 layouts) ----------------
// A (16xK, row-major, 16-bit): lane L<16 holds row L, K = k..k+7 and k+16..k+23;
// lane L>=16 holds row L-16, K = k+8..k+15 and k+24..k+31.
static __device__ __forceinline__ v16h ldA16(const _Float16* A, int lda, int k, int lane) {
  int r  = lane & 15;
  int kb = k + ((lane >> 4) << 3);
  const _Float16* p = A + r * lda + kb;
  v8h lo = *(const v8h*)(p);
  v8h hi = *(const v8h*)(p + 16);
  v16h o;
#pragma unroll
  for (int i = 0; i < 8; ++i) { o[i] = lo[i]; o[i + 8] = hi[i]; }
  return o;
}

// B (Kx16, 16-bit), supplied transposed as Bt[n][k] row-major:
// lane L<16 holds col L, K = k..k+7 and k+16..k+23; lane L>=16 holds col L-16, K = k+8..k+15 and k+24..k+31.
static __device__ __forceinline__ v16h ldB16(const _Float16* Bt, int ldb, int k, int lane) {
  int c  = lane & 15;
  int kb = k + ((lane >> 4) << 3);
  const _Float16* p = Bt + c * ldb + kb;
  v8h lo = *(const v8h*)(p);
  v8h hi = *(const v8h*)(p + 16);
  v16h o;
#pragma unroll
  for (int i = 0; i < 8; ++i) { o[i] = lo[i]; o[i + 8] = hi[i]; }
  return o;
}

// Accumulate C(16x16) += A(16xK) * B(Kx16) over K in steps of 32.
static __device__ __forceinline__ v8f wmma_tile(v8f acc, const _Float16* A, int lda,
                                                const _Float16* Bt, int ldb, int K, int lane) {
  for (int k = 0; k < K; k += 32) {
    v16h a = ldA16(A, lda, k, lane);
    v16h b = ldB16(Bt, ldb, k, lane);
    acc = __builtin_amdgcn_wmma_f32_16x16x32_f16(false, a, false, b, (short)0, acc, false, false);
  }
  return acc;
}

// ---------------- pack / convert kernels (f32 -> f16, transposes) ----------------
__global__ void k_pack_ut(const float* __restrict__ U, _Float16* __restrict__ Ut) {
  int idx = blockIdx.x * blockDim.x + threadIdx.x;          // [B][M][N]
  if (idx >= B_ * M_ * N_) return;
  int b = idx >> 18, r = idx & ((1 << 18) - 1);
  int m = r >> 10, i = r & 1023;
  Ut[idx] = (_Float16)U[((size_t)b << 20) + (size_t)i * N_ + m];
}

__global__ void k_pack_uc(const float* __restrict__ U, _Float16* __restrict__ Uc) {
  int idx = blockIdx.x * blockDim.x + threadIdx.x;          // [B][N][M]
  if (idx >= B_ * N_ * M_) return;
  int b = idx >> 18, r = idx & ((1 << 18) - 1);
  int n = r >> 8, m = r & 255;
  Uc[idx] = (_Float16)U[((size_t)b << 20) + (size_t)n * N_ + m];
}

__global__ void k_pack_x(const float* __restrict__ x, _Float16* __restrict__ xT) {
  int idx = blockIdx.x * blockDim.x + threadIdx.x;          // [B][E][N]
  if (idx >= B_ * E_ * N_) return;
  int b = idx >> 17, r = idx & ((1 << 17) - 1);
  int e = r >> 10, i = r & 1023;
  xT[idx] = (_Float16)x[((size_t)b << 17) + (size_t)i * E_ + e];
}

__global__ void k_pack_w(const float* __restrict__ W, _Float16* __restrict__ Wt) {
  int idx = blockIdx.x * blockDim.x + threadIdx.x;          // [M][E][E], Wt[m][f][e]=W[m][e][f]
  if (idx >= M_ * E_ * E_) return;
  int m = idx >> 14, r = idx & 16383;
  int f = r >> 7, e = r & 127;
  Wt[idx] = (_Float16)W[((size_t)m << 14) + (size_t)e * E_ + f];
}

// ---------------- Stage A: graph Fourier transform (first M modes) ----------------
// Y0[m][b][e] = sum_i U[b][i][m] * x[b][i][e]
__global__ void __launch_bounds__(256) k_gft(const _Float16* __restrict__ Ut,
                                             const _Float16* __restrict__ xT,
                                             float* __restrict__ Y0) {
  int lane = threadIdx.x & 31;
  int tid  = blockIdx.x * 8 + (threadIdx.x >> 5);           // 1024 tiles
  int et = tid & 7, mt = (tid >> 3) & 15, b = tid >> 7;
  const _Float16* A  = Ut + ((size_t)b * M_ + mt * 16) * N_;
  const _Float16* Bt = xT + ((size_t)b * E_ + et * 16) * N_;
  v8f acc = {};
  acc = wmma_tile(acc, A, N_, Bt, N_, N_, lane);
  int rb = (lane >> 4) << 3, col = lane & 15;
#pragma unroll
  for (int j = 0; j < 8; ++j) {
    int m = mt * 16 + rb + j;
    Y0[((size_t)m * B_ + b) * E_ + et * 16 + col] = acc[j];
  }
}

// ---------------- Stage B: RK4 (3/8 rule) per mode ----------------
// One workgroup per mode; 8 waves, each owns a 16-wide column slice of the 16x128 state.
// W[m] (32 KB) is staged into LDS by the Tensor Data Mover (one TDM op from wave 0).
__global__ void __launch_bounds__(256) k_rk4(const float* __restrict__ times,
                                             const _Float16* __restrict__ Wt,
                                             const float* __restrict__ Y0,
                                             _Float16* __restrict__ solT) {
  __shared__ alignas(128) _Float16 Wl[E_ * E_];  // Wt[m] : [f][e]
  __shared__ alignas(32)  _Float16 Yl[16 * E_];  // padded state (rows 8..15 = 0), f16
  int m    = blockIdx.x;
  int lane = threadIdx.x & 31;
  int w    = threadIdx.x >> 5;

  // ---- TDM: DMA W[m] (4096 x 8B = 32 KB, single row) global -> LDS ----
  if (w == 0) {                                  // wave-uniform: only wave 0 issues
    const _Float16* wsrc = Wt + ((size_t)m * E_ * E_);
    unsigned long long gaddr = (unsigned long long)(uintptr_t)wsrc;
    unsigned lds_off = (unsigned)(uintptr_t)(void*)&Wl[0];   // flat[31:0] == LDS offset

    v4u g0;                                      // D# group 0 (ISA 8.3)
    g0[0] = 1u;                                  // count=1, user descriptor
    g0[1] = lds_off;                             // lds_addr
    g0[2] = (unsigned)gaddr;                     // global_addr[31:0]
    g0[3] = (unsigned)((gaddr >> 32) & 0x1FFFFFFu) | (2u << 30); // addr[56:32], type=2

    v8u g1;                                      // D# group 1 (ISA 8.4)
    g1[0] = (3u << 16);                          // workgroup_mask=0, data_size=3 (8B)
    g1[1] = (4096u << 16);                       // tensor_dim0[15:0] = 4096
    g1[2] = (1u << 16);                          // tensor_dim0[31:16]=0, tensor_dim1=1
    g1[3] = (4096u << 16);                       // tile_dim0 = 4096
    g1[4] = 1u;                                  // tile_dim1 = 1, tile_dim2 = 0
    g1[5] = 4096u;                               // tensor_dim0_stride[31:0]
    g1[6] = 0u;
    g1[7] = 0u;

    asm volatile("tensor_load_to_lds %0, %1" :: "s"(g0), "s"(g1) : "memory");
    __builtin_amdgcn_s_wait_tensorcnt(0);        // TENSORcnt==0 before barrier signal
  }

  // ---- init state (runs concurrently with the TDM fill) ----
  int rb  = (lane >> 4) << 3;
  int col = (lane & 15) + w * 16;
  v8f y;
#pragma unroll
  for (int j = 0; j < 8; ++j) {
    int r = rb + j;
    float v = (r < B_) ? Y0[((size_t)m * B_ + r) * E_ + col] : 0.0f;
    y[j] = v;
    Yl[r * E_ + col] = (_Float16)v;
  }
  __syncthreads();

  const _Float16* Btile = Wl + (w * 16) * E_;    // this wave's 16 output channels
  const float third = 1.0f / 3.0f;
  float tprev = 0.0f;

  for (int t = 0; t < T_; ++t) {
    float tcur = times[t];                        // shared time grid (row 0)
    float h = tcur - tprev; tprev = tcur;

    v8f z = {};
    v8f k1 = wmma_tile(z, Yl, E_, Btile, E_, E_, lane);
    __syncthreads();
#pragma unroll
    for (int j = 0; j < 8; ++j) Yl[(rb + j) * E_ + col] = (_Float16)(y[j] + h * third * k1[j]);
    __syncthreads();

    v8f k2 = wmma_tile(z, Yl, E_, Btile, E_, E_, lane);
    __syncthreads();
#pragma unroll
    for (int j = 0; j < 8; ++j) Yl[(rb + j) * E_ + col] = (_Float16)(y[j] + h * (k2[j] - third * k1[j]));
    __syncthreads();

    v8f k3 = wmma_tile(z, Yl, E_, Btile, E_, E_, lane);
    __syncthreads();
#pragma unroll
    for (int j = 0; j < 8; ++j) Yl[(rb + j) * E_ + col] = (_Float16)(y[j] + h * (k1[j] - k2[j] + k3[j]));
    __syncthreads();

    v8f k4 = wmma_tile(z, Yl, E_, Btile, E_, E_, lane);
    __syncthreads();
#pragma unroll
    for (int j = 0; j < 8; ++j) {
      y[j] = y[j] + h * 0.125f * (k1[j] + 3.0f * (k2[j] + k3[j]) + k4[j]);
      Yl[(rb + j) * E_ + col] = (_Float16)y[j];
      int b = rb + j;
      if (b < B_) {                               // solT[b][t][e][m]  (Bt for stage C)
        solT[(((size_t)b * T_ + t) * E_ + col) * M_ + m] = (_Float16)y[j];
      }
    }
    __syncthreads();
  }
}

// ---------------- Stage C: inverse GFT ----------------
// out[t][b][n][e] = sum_m U[b][n][m] * y_t[m][b][e]
__global__ void __launch_bounds__(256) k_igft(const _Float16* __restrict__ Uc,
                                              const _Float16* __restrict__ solT,
                                              float* __restrict__ out) {
  int lane = threadIdx.x & 31;
  int tid  = blockIdx.x * 8 + (threadIdx.x >> 5);           // 65536 tiles
  int et = tid & 7;
  int nt = (tid >> 3) & 63;
  int b  = (tid >> 9) & 7;
  int t  = tid >> 12;
  const _Float16* A  = Uc + ((size_t)b * N_ + nt * 16) * M_;
  const _Float16* Bt = solT + (((size_t)b * T_ + t) * E_ + et * 16) * M_;
  __builtin_prefetch(A, 0, 3);
  __builtin_prefetch(Bt, 0, 3);
  v8f acc = {};
  acc = wmma_tile(acc, A, M_, Bt, M_, M_, lane);
  int rb = (lane >> 4) << 3, col = lane & 15;
#pragma unroll
  for (int j = 0; j < 8; ++j) {
    int n = nt * 16 + rb + j;
    out[(((size_t)t * B_ + b) * N_ + n) * E_ + et * 16 + col] = acc[j];
  }
}

// ---------------- host-side launch ----------------
extern "C" void kernel_launch(void* const* d_in, const int* in_sizes, int n_in,
                              void* d_out, int out_size, void* d_ws, size_t ws_size,
                              hipStream_t stream) {
  const float* x     = (const float*)d_in[0];   // [B,N,E]
  const float* times = (const float*)d_in[1];   // [B,T]
  const float* U     = (const float*)d_in[2];   // [B,N,N]
  const float* W     = (const float*)d_in[3];   // [M,E,E]
  float* out = (float*)d_out;                   // [T,B,N,E]

  char* ws = (char*)d_ws;
  _Float16* Ut   = (_Float16*)(ws);                         // 4 MB  [B][M][N]
  _Float16* Uc   = (_Float16*)(ws + (4  << 20));            // 4 MB  [B][N][M]
  _Float16* xT   = (_Float16*)(ws + (8  << 20));            // 2 MB  [B][E][N]
  _Float16* Wt   = (_Float16*)(ws + (10 << 20));            // 8 MB  [M][E][E]
  _Float16* solT = (_Float16*)(ws + (18 << 20));            // 8 MB  [B][T][E][M]
  float*    Y0   = (float*)   (ws + (26 << 20));            // 1 MB  [M][B][E]

  (void)in_sizes; (void)n_in; (void)out_size; (void)ws_size;

  k_pack_ut<<<(B_ * M_ * N_ + 255) / 256, 256, 0, stream>>>(U, Ut);
  k_pack_uc<<<(B_ * N_ * M_ + 255) / 256, 256, 0, stream>>>(U, Uc);
  k_pack_x <<<(B_ * E_ * N_ + 255) / 256, 256, 0, stream>>>(x, xT);
  k_pack_w <<<(M_ * E_ * E_ + 255) / 256, 256, 0, stream>>>(W, Wt);

  k_gft <<<128, 256, 0, stream>>>(Ut, xT, Y0);              // 1024 tiles
  k_rk4 <<<M_, 256, 0, stream>>>(times, Wt, Y0, solT);      // one block per mode
  k_igft<<<8192, 256, 0, stream>>>(Uc, solT, out);          // 65536 tiles
}
